// MoEKGC_21328807592497
// MI455X (gfx1250) — matrix operand order repaired and checked by reading, not compiled, not run on hardware
//
#include <hip/hip_runtime.h>
#include <hip/hip_bf16.h>
#include <math.h>
#include <stdint.h>

#define N_NODES 50000
#define E_DIM   256
#define H_DIM   512
#define NEXP    5
#define GH      128
#define NEDGES  800000
#define NCOLS   512           // every big GEMM has 512 output columns

typedef __attribute__((ext_vector_type(16))) __bf16 v16bf;
typedef __attribute__((ext_vector_type(8)))  float  v8f;

union Frag { v16bf v; uint4 q[2]; };

#define M_TILE       32
#define K_PANEL      64
#define AS_STRIDE    72       // 144B rows: 16B-aligned frag reads, 36-bank stride
#define BT_STRIDE    72
#define A_LDS_BYTES  (M_TILE * AS_STRIDE * 2)                        // 4608
#define PANEL_BYTES  ((M_TILE*AS_STRIDE + NCOLS*BT_STRIDE) * 2)      // 78336
#define GEMM_THREADS 256
#define GEMM_SMEM    (2 * PANEL_BYTES)                               // ping-pong

__device__ __forceinline__ float gelu_tanh(float x) {
    float x3 = x * x * x;
    return 0.5f * x * (1.0f + tanhf(0.7978845608028654f * (x + 0.044715f * x3)));
}

// Async DMA of one 16-byte chunk global -> LDS (gfx1250 ASYNCcnt path).
__device__ __forceinline__ void async_g2l_b128(unsigned lds_byte_off, const void* gaddr) {
    asm volatile("global_load_async_to_lds_b128 %0, %1, off"
                 :: "v"(lds_byte_off), "v"((unsigned long long)gaddr) : "memory");
}
__device__ __forceinline__ void wait_async0() {
    asm volatile("s_wait_asynccnt 0x0" ::: "memory");
}

// C[M x 512] = act(A[M x K] * Bt^T + bias + addend)
// A: bf16 row-major [M x K].  Bt: bf16 [512 x K] (pre-transposed weights).
// Ping-pong LDS panels staged by async DMA, overlapped with WMMA compute.
// Fragment loads are a single asm block: 18 ds_load_b128 in flight, ONE
// s_wait_dscnt, then 8 back-to-back WMMAs (defeats per-load wait scheduling).
__global__ __launch_bounds__(GEMM_THREADS)
__attribute__((amdgpu_waves_per_eu(2, 2)))
void gemm512_wmma(const __bf16* __restrict__ A, int M, int K,
                  const __bf16* __restrict__ Bt,
                  const float* __restrict__ bias,     // [512] or null
                  const float* __restrict__ addend,   // [M x 512] or null
                  int act,                            // 0=none 1=relu 2=gelu
                  float*  __restrict__ outF,          // f32 out or null
                  __bf16* __restrict__ outB,          // bf16 out or null
                  int gated,
                  const float* __restrict__ gates,    // [M x 2]
                  const int*   __restrict__ idx2,     // [M x 2]
                  int expert)
{
    extern __shared__ __bf16 smem[];
    const unsigned lds_base = (unsigned)(uintptr_t)smem;  // LDS byte offset (low 32b)

    const int tid   = threadIdx.x;
    const int wave  = tid >> 5;
    const int lane  = tid & 31;
    const int lhalf = lane >> 4;      // 0/1
    const int l15   = lane & 15;
    const int rtile = wave >> 2;      // 0..1  (16-row tile)
    const int cgrp  = wave & 3;       // 0..3  (8 col-tiles each)
    const int row0  = blockIdx.x * M_TILE;

    // A staging task (1 chunk/thread): row r, 16B chunk ch within 64 cols
    const int a_r  = tid >> 3;        // 0..31
    const int a_ch = tid & 7;
    int a_gr = row0 + a_r; if (a_gr > M - 1) a_gr = M - 1;   // clamp tail (rows discarded)

    auto stage = [&](int k0, unsigned poff) {
        async_g2l_b128(lds_base + poff + (unsigned)(a_r * (AS_STRIDE * 2) + a_ch * 16),
                       A + (size_t)a_gr * K + k0 + a_ch * 8);
        #pragma unroll
        for (int t = 0; t < 16; ++t) {
            int task = tid + t * GEMM_THREADS;   // 0..4095
            int c  = task >> 3;
            int ch = task & 7;
            unsigned lb = lds_base + poff +
                          (unsigned)(A_LDS_BYTES + c * (BT_STRIDE * 2) + ch * 16);
            async_g2l_b128(lb, Bt + (size_t)c * K + k0 + ch * 8);
        }
    };

    // per-lane fragment base offsets (bytes, within a panel)
    const unsigned aFragBase = (unsigned)(((rtile * 16 + l15) * AS_STRIDE + 16 * lhalf) * 2);
    const unsigned bFragBase = (unsigned)(A_LDS_BYTES +
                               ((cgrp * 128 + l15) * BT_STRIDE + 16 * lhalf) * 2);

    v8f acc[8] = {};

    // prologue: stage first panel
    stage(0, 0);
    wait_async0();
    __syncthreads();

    unsigned pp = 0;
    for (int k0 = 0; k0 < K; k0 += K_PANEL) {
        const unsigned cur = pp;
        const unsigned nxt = pp ^ PANEL_BYTES;
        if (k0 + K_PANEL < K) stage(k0 + K_PANEL, nxt);   // DMA overlaps compute below

        const unsigned lds_cur = lds_base + cur;

        #pragma unroll
        for (int kk = 0; kk < K_PANEL; kk += 32) {
            unsigned aAddr = lds_cur + aFragBase + (unsigned)(kk * 2);
            unsigned bAddr = lds_cur + bFragBase + (unsigned)(kk * 2);
            Frag af, bfr[8];
            // 18 ds_load_b128 in flight, single wait inside the asm block.
            // B frag i at bAddr + i*2304 (= 16*BT_STRIDE*2); q1 at +16 bytes.
            asm volatile(
                "ds_load_b128 %0, %18\n\t"
                "ds_load_b128 %1, %18 offset:16\n\t"
                "ds_load_b128 %2, %19\n\t"
                "ds_load_b128 %3, %19 offset:16\n\t"
                "ds_load_b128 %4, %19 offset:2304\n\t"
                "ds_load_b128 %5, %19 offset:2320\n\t"
                "ds_load_b128 %6, %19 offset:4608\n\t"
                "ds_load_b128 %7, %19 offset:4624\n\t"
                "ds_load_b128 %8, %19 offset:6912\n\t"
                "ds_load_b128 %9, %19 offset:6928\n\t"
                "ds_load_b128 %10, %19 offset:9216\n\t"
                "ds_load_b128 %11, %19 offset:9232\n\t"
                "ds_load_b128 %12, %19 offset:11520\n\t"
                "ds_load_b128 %13, %19 offset:11536\n\t"
                "ds_load_b128 %14, %19 offset:13824\n\t"
                "ds_load_b128 %15, %19 offset:13840\n\t"
                "ds_load_b128 %16, %19 offset:16128\n\t"
                "ds_load_b128 %17, %19 offset:16144\n\t"
                "s_wait_dscnt 0x0"
                : "=v"(af.q[0]),     "=v"(af.q[1]),
                  "=v"(bfr[0].q[0]), "=v"(bfr[0].q[1]),
                  "=v"(bfr[1].q[0]), "=v"(bfr[1].q[1]),
                  "=v"(bfr[2].q[0]), "=v"(bfr[2].q[1]),
                  "=v"(bfr[3].q[0]), "=v"(bfr[3].q[1]),
                  "=v"(bfr[4].q[0]), "=v"(bfr[4].q[1]),
                  "=v"(bfr[5].q[0]), "=v"(bfr[5].q[1]),
                  "=v"(bfr[6].q[0]), "=v"(bfr[6].q[1]),
                  "=v"(bfr[7].q[0]), "=v"(bfr[7].q[1])
                : "v"(aAddr), "v"(bAddr)
                : "memory");
            #pragma unroll
            for (int i = 0; i < 8; ++i) {
                acc[i] = __builtin_amdgcn_wmma_f32_16x16x32_bf16(
                    false, af.v, false, bfr[i].v, (short)0, acc[i], false, false);
            }
        }

        wait_async0();        // next panel's DMA (mostly) done by now
        __syncthreads();
        pp = nxt;
    }

    // ---- epilogue: C layout: VGPR vi -> M = vi + 8*lhalf, N = ct*16 + (lane&15)
    #pragma unroll
    for (int i = 0; i < 8; ++i) {
        int n = (cgrp * 8 + i) * 16 + l15;
        float bval = bias ? bias[n] : 0.f;
        #pragma unroll
        for (int vi = 0; vi < 8; ++vi) {
            int m  = vi + 8 * lhalf;
            int gr = row0 + rtile * 16 + m;
            if (gr < M) {
                float v = acc[i][vi] + bval;
                if (addend) v += addend[(size_t)gr * NCOLS + n];
                if (act == 1)      v = fmaxf(v, 0.f);
                else if (act == 2) v = gelu_tanh(v);
                if (gated) {
                    float w = 0.f;
                    int i0 = idx2[gr * 2], i1 = idx2[gr * 2 + 1];
                    if (i0 == expert)      w = gates[gr * 2];
                    else if (i1 == expert) w = gates[gr * 2 + 1];
                    if (w != 0.f)
                        unsafeAtomicAdd(&outF[(size_t)gr * NCOLS + n], w * v);
                } else {
                    if (outF) outF[(size_t)gr * NCOLS + n] = v;
                    if (outB) outB[(size_t)gr * NCOLS + n] = (__bf16)v;
                }
            }
        }
    }
}

// fp32 -> bf16 elementwise (n multiple of 4)
__global__ __launch_bounds__(256)
void cvt_bf16_kernel(const float* __restrict__ in, __bf16* __restrict__ out, size_t n)
{
    size_t i = ((size_t)blockIdx.x * 256 + threadIdx.x) * 4;
    if (i >= n) return;
    float4 v = *reinterpret_cast<const float4*>(in + i);
    ushort4 pk;
    pk.x = __builtin_bit_cast(unsigned short, (__bf16)v.x);
    pk.y = __builtin_bit_cast(unsigned short, (__bf16)v.y);
    pk.z = __builtin_bit_cast(unsigned short, (__bf16)v.z);
    pk.w = __builtin_bit_cast(unsigned short, (__bf16)v.w);
    *reinterpret_cast<ushort4*>(out + i) = pk;
}

// weight pre-transpose + convert: in [K x 512] f32 -> out [512 x K] bf16
__global__ __launch_bounds__(256)
void transpose_cvt_kernel(const float* __restrict__ in, __bf16* __restrict__ out, int K)
{
    __shared__ float tile[32][33];
    const int tx = threadIdx.x & 31, ty = threadIdx.x >> 5;   // 32 x 8
    const int c0 = blockIdx.x * 32;                           // along 512
    const int k0 = blockIdx.y * 32;                           // along K
    #pragma unroll
    for (int i = 0; i < 32; i += 8)
        tile[ty + i][tx] = in[(size_t)(k0 + ty + i) * NCOLS + c0 + tx];
    __syncthreads();
    #pragma unroll
    for (int i = 0; i < 32; i += 8)
        out[(size_t)(c0 + ty + i) * K + k0 + tx] = (__bf16)tile[tx][ty + i];
}

// Gating network in full fp32 (numerically sensitive: drives top-k routing).
#define GATE_THREADS 128
__global__ __launch_bounds__(GATE_THREADS)
void gating_kernel(const float* __restrict__ text,
                   const float* __restrict__ tabular,
                   const float* __restrict__ structured,
                   const float* __restrict__ Wg1, const float* __restrict__ bg1,
                   const float* __restrict__ Wg2, const float* __restrict__ bg2,
                   float* __restrict__ gates, int* __restrict__ idx2)
{
    __shared__ float xs[16][3 * E_DIM];   // 48 KB
    __shared__ float gs[16][GH];          //  8 KB
    const int tid   = threadIdx.x;
    const int node0 = blockIdx.x * 16;
    const float* seg[3] = {text, tabular, structured};

    #pragma unroll
    for (int t = 0; t < 24; ++t) {
        int task = tid + t * GATE_THREADS;      // 0..3071 float4 tasks
        int r = task / 192;
        int q = task % 192;
        int s = q >> 6;
        int off = (q & 63) * 4;
        float4 v = *reinterpret_cast<const float4*>(seg[s] + (size_t)(node0 + r) * E_DIM + off);
        *reinterpret_cast<float4*>(&xs[r][s * E_DIM + off]) = v;
    }
    __syncthreads();

    {
        int j = tid;
        float acc[16];
        #pragma unroll
        for (int r = 0; r < 16; ++r) acc[r] = bg1[j];
        for (int k = 0; k < 3 * E_DIM; ++k) {
            float w = Wg1[k * GH + j];
            #pragma unroll
            for (int r = 0; r < 16; ++r) acc[r] += xs[r][k] * w;
        }
        #pragma unroll
        for (int r = 0; r < 16; ++r) gs[r][j] = fmaxf(acc[r], 0.f);
    }
    __syncthreads();

    if (tid < 16) {
        int r = tid;
        float lg[NEXP];
        #pragma unroll
        for (int c = 0; c < NEXP; ++c) lg[c] = bg2[c];
        for (int j = 0; j < GH; ++j) {
            float g = gs[r][j];
            #pragma unroll
            for (int c = 0; c < NEXP; ++c) lg[c] += g * Wg2[j * NEXP + c];
        }
        float m = lg[0];
        #pragma unroll
        for (int c = 1; c < NEXP; ++c) m = fmaxf(m, lg[c]);
        float ex[NEXP];
        #pragma unroll
        for (int c = 0; c < NEXP; ++c) ex[c] = __expf(lg[c] - m);
        int i0 = 0; float v0 = ex[0];
        #pragma unroll
        for (int c = 1; c < NEXP; ++c) if (ex[c] > v0) { v0 = ex[c]; i0 = c; }
        int i1 = -1; float v1 = -1.f;
        #pragma unroll
        for (int c = 0; c < NEXP; ++c) if (c != i0 && ex[c] > v1) { v1 = ex[c]; i1 = c; }
        float s = v0 + v1;
        int node = node0 + r;
        gates[node * 2]     = v0 / s;
        gates[node * 2 + 1] = v1 / s;
        idx2[node * 2]      = i0;
        idx2[node * 2 + 1]  = i1;
    }
}

// msg[dst] += tmp[src] over all edges; 1 thread = 1 edge x 4 columns.
__global__ __launch_bounds__(256)
void scatter_kernel(const int* __restrict__ ei,
                    const float* __restrict__ tmp,
                    float* __restrict__ msg)
{
    size_t gid = (size_t)blockIdx.x * 256 + threadIdx.x;
    if (gid >= (size_t)NEDGES * (H_DIM / 4)) return;
    int e = (int)(gid >> 7);
    int c = (int)(gid & 127) * 4;
    int src = ei[e];
    int dst = ei[NEDGES + e];
    float4 v = *reinterpret_cast<const float4*>(tmp + (size_t)src * H_DIM + c);
    float* d = msg + (size_t)dst * H_DIM + c;
    unsafeAtomicAdd(d + 0, v.x);
    unsafeAtomicAdd(d + 1, v.y);
    unsafeAtomicAdd(d + 2, v.z);
    unsafeAtomicAdd(d + 3, v.w);
}

extern "C" void kernel_launch(void* const* d_in, const int* in_sizes, int n_in,
                              void* d_out, int out_size, void* d_ws, size_t ws_size,
                              hipStream_t stream)
{
    const float* text       = (const float*)d_in[0];
    const float* tabular    = (const float*)d_in[1];
    const float* structured = (const float*)d_in[2];
    const int*   edge_index = (const int*)d_in[3];
    const float* Wg1  = (const float*)d_in[4];
    const float* bg1  = (const float*)d_in[5];
    const float* Wg2  = (const float*)d_in[6];
    const float* bg2  = (const float*)d_in[7];
    const float* We1  = (const float*)d_in[8];
    const float* be1  = (const float*)d_in[9];
    const float* We2  = (const float*)d_in[10];
    const float* be2  = (const float*)d_in[11];
    const float* Wself = (const float*)d_in[12];
    const float* Wnbr  = (const float*)d_in[13];
    float* out = (float*)d_out;

    char* ws = (char*)d_ws;
    size_t off = 0;
    auto alloc = [&](size_t bytes) -> char* {
        char* p = ws + off;
        off += (bytes + 255) & ~(size_t)255;
        return p;
    };
    const size_t NH = (size_t)N_NODES * H_DIM;
    const size_t NE_ = (size_t)N_NODES * E_DIM;
    float*  gates = (float*)alloc((size_t)N_NODES * 2 * sizeof(float));
    int*    idx2  = (int*)  alloc((size_t)N_NODES * 2 * sizeof(int));
    float*  nf    = (float*)alloc(NH * sizeof(float));   // node feats / h (f32)
    float*  tmp   = (float*)alloc(NH * sizeof(float));   // GNN neighbor GEMM out
    float*  msg   = (float*)alloc(NH * sizeof(float));   // scatter accumulator
    __bf16* hbf   = (__bf16*)alloc(NH * sizeof(__bf16)); // h (bf16 A operand)
    __bf16* xin_bf[3];
    for (int s = 0; s < 3; ++s) xin_bf[s] = (__bf16*)alloc(NE_ * sizeof(__bf16));
    __bf16* We1t  = (__bf16*)alloc((size_t)NEXP * NCOLS * E_DIM * sizeof(__bf16));
    __bf16* We2t  = (__bf16*)alloc((size_t)NEXP * NCOLS * H_DIM * sizeof(__bf16));
    __bf16* Wselft= (__bf16*)alloc((size_t)2 * NCOLS * H_DIM * sizeof(__bf16));
    __bf16* Wnbrt = (__bf16*)alloc((size_t)2 * NCOLS * H_DIM * sizeof(__bf16));
    __bf16* h1bf  = (__bf16*)tmp;                        // alias: h1 (bf16) lives in tmp

    hipMemsetAsync(nf, 0, NH * sizeof(float), stream);

    // one-time conversions (cheap: ~160 MB total traffic vs 23.3 TB/s)
    const float* xin_src[3] = {text, tabular, structured};
    for (int s = 0; s < 3; ++s)
        cvt_bf16_kernel<<<(int)(NE_ / 4 + 255) / 256, 256, 0, stream>>>(xin_src[s], xin_bf[s], NE_);
    for (int e = 0; e < NEXP; ++e) {
        transpose_cvt_kernel<<<dim3(16, E_DIM / 32), 256, 0, stream>>>(
            We1 + (size_t)e * E_DIM * NCOLS, We1t + (size_t)e * NCOLS * E_DIM, E_DIM);
        transpose_cvt_kernel<<<dim3(16, H_DIM / 32), 256, 0, stream>>>(
            We2 + (size_t)e * H_DIM * NCOLS, We2t + (size_t)e * NCOLS * H_DIM, H_DIM);
    }
    for (int l = 0; l < 2; ++l) {
        transpose_cvt_kernel<<<dim3(16, H_DIM / 32), 256, 0, stream>>>(
            Wself + (size_t)l * H_DIM * NCOLS, Wselft + (size_t)l * NCOLS * H_DIM, H_DIM);
        transpose_cvt_kernel<<<dim3(16, H_DIM / 32), 256, 0, stream>>>(
            Wnbr + (size_t)l * H_DIM * NCOLS, Wnbrt + (size_t)l * NCOLS * H_DIM, H_DIM);
    }

    gating_kernel<<<N_NODES / 16, GATE_THREADS, 0, stream>>>(
        text, tabular, structured, Wg1, bg1, Wg2, bg2, gates, idx2);

    // xin stack order: [tabular, structured, text, text, structured]
    const __bf16* xin_e[NEXP] = {xin_bf[1], xin_bf[2], xin_bf[0], xin_bf[0], xin_bf[2]};
    const int gblocks = (N_NODES + M_TILE - 1) / M_TILE;

    for (int e = 0; e < NEXP; ++e) {
        // h1 = gelu(xin_e @ We1[e] + be1[e])  -> bf16
        gemm512_wmma<<<gblocks, GEMM_THREADS, GEMM_SMEM, stream>>>(
            xin_e[e], N_NODES, E_DIM, We1t + (size_t)e * NCOLS * E_DIM,
            be1 + (size_t)e * H_DIM, nullptr, /*act=*/2,
            nullptr, h1bf, /*gated=*/0, nullptr, nullptr, 0);
        // nf += gate_e * (h1 @ We2[e] + be2[e])
        gemm512_wmma<<<gblocks, GEMM_THREADS, GEMM_SMEM, stream>>>(
            h1bf, N_NODES, H_DIM, We2t + (size_t)e * NCOLS * H_DIM,
            be2 + (size_t)e * H_DIM, nullptr, /*act=*/0,
            nf, nullptr, /*gated=*/1, gates, idx2, e);
    }

    // h (bf16) for the GNN GEMMs
    cvt_bf16_kernel<<<(int)(NH / 4 + 255) / 256, 256, 0, stream>>>(nf, hbf, NH);

    const size_t sthreads = (size_t)NEDGES * (H_DIM / 4);
    const int sblocks = (int)((sthreads + 255) / 256);

    for (int l = 0; l < 2; ++l) {
        // tmp = h @ Wnbr[l]
        gemm512_wmma<<<gblocks, GEMM_THREADS, GEMM_SMEM, stream>>>(
            hbf, N_NODES, H_DIM, Wnbrt + (size_t)l * NCOLS * H_DIM,
            nullptr, nullptr, /*act=*/0, tmp, nullptr, 0, nullptr, nullptr, 0);
        hipMemsetAsync(msg, 0, NH * sizeof(float), stream);
        scatter_kernel<<<sblocks, 256, 0, stream>>>(edge_index, tmp, msg);
        // h = relu(h @ Wself[l] + msg); in-place safe (block reads only its own rows)
        gemm512_wmma<<<gblocks, GEMM_THREADS, GEMM_SMEM, stream>>>(
            hbf, N_NODES, H_DIM, Wselft + (size_t)l * NCOLS * H_DIM,
            nullptr, msg, /*act=*/1, (l == 1 ? out : nf), hbf, 0, nullptr, nullptr, 0);
    }

    (void)in_sizes; (void)n_in; (void)out_size; (void)ws_size;
}
